// TransformerDecoderLayer_65094524338339
// MI455X (gfx1250) — compile-verified, matrix operand order checked by reading, not run on hardware
//
#include <hip/hip_runtime.h>
#include <hip/hip_bf16.h>
#include <math.h>
#include <stdint.h>

typedef __bf16 bf16;
typedef __attribute__((ext_vector_type(16))) __bf16 v16bf;
typedef __attribute__((ext_vector_type(8)))  float   v8f;

#define WMMA_BF16(a, b, c) \
  __builtin_amdgcn_wmma_f32_16x16x32_bf16(false, (a), false, (b), (short)0, (c), false, false)

namespace {
constexpr int D   = 256;
constexpr int H   = 8;
constexpr int Q   = 12;
constexpr int S   = 512;
constexpr int B   = 256;
constexpr int DFF = 2048;
constexpr int QB  = Q * B;   // 3072
constexpr int SB  = S * B;   // 131072
}

// v16bf fragment element -> K index inside the 32-wide K tile (ISA 7.12.2)
__device__ __forceinline__ int kmap(int e, int lane) {
  return (e & 7) + ((e >> 3) << 4) + ((lane & 16) >> 1);
}

// ---------------------------------------------------------------------------
// Generic bf16 WMMA GEMM: C = act((A[MxK] * W[KxN] + bias) * alpha)
// Block tile 128x64, 8 waves, each wave 32x32 (2x2 WMMA tiles), K step 32.
// A tile staged via async global->LDS DMA (ASYNCcnt); W tile transposed in LDS.
// ---------------------------------------------------------------------------
__global__ __launch_bounds__(256) void gemm_bf16_kernel(
    const bf16* __restrict__ A, const bf16* __restrict__ W,
    const float* __restrict__ bias, float* Cf, bf16* Cb,
    int M, int N, int K, float alpha, int relu, int accum)
{
  __shared__ bf16 As[128][40];   // padded; row stride 80B (16B-aligned)
  __shared__ bf16 Bs[64][40];    // stored transposed: Bs[n][k]

  const int tid  = threadIdx.x;
  const int lane = tid & 31;
  const int wave = tid >> 5;
  const int m0 = blockIdx.y * 128;
  const int n0 = blockIdx.x * 64;
  const int wm = (wave & 3) * 32;
  const int wn = (wave >> 2) * 32;

  const int arow = tid >> 1;            // A staging: 128 rows x 2 threads
  const int acb  = (tid & 1) * 16;      // 16 bf16 (32B) per thread
  const int krow = tid >> 3;            // W staging: 32 k-rows x 8 threads
  const int wcb  = (tid & 7) * 8;       // 8 bf16 per thread
  const bool n_full = (n0 + 64 <= N);   // block-uniform fast path

  v8f acc[2][2] = {};

  for (int kt = 0; kt < K; kt += 32) {
    // ---- A tile (128x32): async copy straight into LDS ----
    {
      int m = m0 + arow;
      unsigned lds = (unsigned)(uintptr_t)&As[arow][acb];
      if (m < M) {
        const bf16* src = A + (size_t)m * K + kt + acb;
        asm volatile(
            "global_load_async_to_lds_b128 %0, %1, off\n\t"
            "global_load_async_to_lds_b128 %0, %1, off offset:16"
            :: "v"(lds), "v"(src) : "memory");
      } else {
        #pragma unroll
        for (int e = 0; e < 16; ++e) As[arow][acb + e] = (bf16)0.f;
      }
    }
    // ---- W tile (32x64), transposed into LDS ----
    if (n_full) {
      const bf16* src = W + (size_t)(kt + krow) * N + n0 + wcb;
      uint4 u = *(const uint4*)src;
      const bf16* p = (const bf16*)&u;
      #pragma unroll
      for (int e = 0; e < 8; ++e) Bs[wcb + e][krow] = p[e];
    } else {
      #pragma unroll
      for (int e = 0; e < 8; ++e) {
        int n = wcb + e;
        bf16 v = (bf16)0.f;
        if (n0 + n < N) v = W[(size_t)(kt + krow) * N + n0 + n];
        Bs[n][krow] = v;
      }
    }
    // ---- prefetch next K tile ----
    if (kt + 32 < K) {
      int m = m0 + arow;
      if (m < M) __builtin_prefetch(A + (size_t)m * K + kt + 32 + acb, 0, 1);
      __builtin_prefetch(W + (size_t)(kt + 32 + krow) * N + n0 + wcb, 0, 1);
    }
    asm volatile("s_wait_asynccnt 0x0" ::: "memory");
    __syncthreads();

    v16bf afr[2], bfr[2];
    #pragma unroll
    for (int s = 0; s < 2; ++s) {
      int rrow = wm + s * 16 + (lane & 15);
      int ncol = wn + s * 16 + (lane & 15);
      #pragma unroll
      for (int e = 0; e < 16; ++e) {
        int kk = kmap(e, lane);
        afr[s][e] = As[rrow][kk];
        bfr[s][e] = Bs[ncol][kk];
      }
    }
    acc[0][0] = WMMA_BF16(afr[0], bfr[0], acc[0][0]);
    acc[0][1] = WMMA_BF16(afr[0], bfr[1], acc[0][1]);
    acc[1][0] = WMMA_BF16(afr[1], bfr[0], acc[1][0]);
    acc[1][1] = WMMA_BF16(afr[1], bfr[1], acc[1][1]);
    __syncthreads();
  }

  // ---- epilogue ----
  #pragma unroll
  for (int i = 0; i < 2; ++i)
    #pragma unroll
    for (int j = 0; j < 2; ++j)
      #pragma unroll
      for (int v = 0; v < 8; ++v) {
        int m = m0 + wm + i * 16 + v + ((lane >> 4) << 3);
        int n = n0 + wn + j * 16 + (lane & 15);
        if (m < M && n < N) {
          float val = acc[i][j][v] + (bias ? bias[n] : 0.f);
          val *= alpha;
          if (relu) val = fmaxf(val, 0.f);
          size_t o = (size_t)m * N + n;
          if (Cf) { if (accum) Cf[o] += val; else Cf[o] = val; }
          if (Cb) Cb[o] = (bf16)val;
        }
      }
}

// ---------------------------------------------------------------------------
// Fused cross-attention (one block per (b,h)), WMMA logits + softmax + WMMA PV
// qh3: (Q,B,768) bf16 pre-scaled, kh3: (S,B,768) bf16, vh: (S,B,256) bf16
// spp/scl: (B*H, Q, 3) f32.  out: (Q,B,256) f32
// ---------------------------------------------------------------------------
__global__ __launch_bounds__(128) void cross_attn_kernel(
    const bf16* __restrict__ qh3, const bf16* __restrict__ kh3,
    const bf16* __restrict__ vh, const float* __restrict__ spp,
    const float* __restrict__ scl, float* __restrict__ out)
{
  __shared__ bf16  Qs[16][96];
  __shared__ bf16  KsW[4][16][96];
  __shared__ float Lg[16][512];
  __shared__ float Sp[16][3];
  __shared__ float Sc[16][3];

  const int bh = blockIdx.x;
  const int b = bh >> 3;
  const int h = bh & 7;
  const int tid  = threadIdx.x;
  const int lane = tid & 31;
  const int wave = tid >> 5;

  for (int idx = tid; idx < 16 * 96; idx += 128) {
    int r = idx / 96, c = idx % 96;
    Qs[r][c] = (r < Q) ? qh3[((size_t)(r * B + b)) * 768 + h * 96 + c] : (bf16)0.f;
  }
  if (tid < 48) {
    int q = tid / 3, i = tid % 3;
    float sv = 0.f, cv = 0.f;
    if (q < Q) {
      size_t o = ((size_t)(b * H + h) * Q + q) * 3 + i;
      sv = spp[o]; cv = scl[o];
    }
    Sp[q][i] = sv; Sc[q][i] = cv;
  }
  __syncthreads();

  v16bf aQ[3];
  #pragma unroll
  for (int ks = 0; ks < 3; ++ks)
    #pragma unroll
    for (int e = 0; e < 16; ++e)
      aQ[ks][e] = Qs[lane & 15][ks * 32 + kmap(e, lane)];

  // ---- logits: 32 s-tiles of 16, round-robin over 4 waves ----
  for (int st = wave; st < 32; st += 4) {
    {
      int r  = lane >> 1;
      int cb = (lane & 1) * 48;
      const bf16* src = kh3 + ((size_t)((st * 16 + r) * B + b)) * 768 + h * 96 + cb;
      #pragma unroll
      for (int e = 0; e < 48; ++e) KsW[wave][r][cb + e] = src[e];
    }
    __syncthreads();
    v8f acc = {};
    #pragma unroll
    for (int ks = 0; ks < 3; ++ks) {
      v16bf bK;
      #pragma unroll
      for (int e = 0; e < 16; ++e)
        bK[e] = KsW[wave][lane & 15][ks * 32 + kmap(e, lane)];
      acc = WMMA_BF16(aQ[ks], bK, acc);
    }
    #pragma unroll
    for (int v = 0; v < 8; ++v) {
      int qr = v + ((lane >> 4) << 3);
      int sc = st * 16 + (lane & 15);
      // gaussian map (sum of squares is >= 0, so -|x|/8 == -x/8)
      float kx = (float)(sc & 7) + 0.5f;
      float ky = (float)((sc >> 3) & 7) + 0.5f;
      float kz = (float)((sc >> 6) & 7) + 0.5f;
      float rx = kx - Sp[qr][0], ry = ky - Sp[qr][1], rz = kz - Sp[qr][2];
      float g = -(rx * rx * Sc[qr][0] * Sc[qr][0] +
                  ry * ry * Sc[qr][1] * Sc[qr][1] +
                  rz * rz * Sc[qr][2] * Sc[qr][2]) * 0.125f;
      Lg[qr][sc] = acc[v] + g;
    }
    __syncthreads();
  }
  __syncthreads();

  // ---- softmax over S; padded rows zeroed; normalized probs stored in Lg ----
  for (int r = wave; r < 16; r += 4) {
    if (r >= Q) {
      for (int c = lane; c < 512; c += 32) Lg[r][c] = 0.f;
      continue;
    }
    float mx = -1e30f;
    for (int c = lane; c < 512; c += 32) mx = fmaxf(mx, Lg[r][c]);
    #pragma unroll
    for (int o = 16; o > 0; o >>= 1) mx = fmaxf(mx, __shfl_xor(mx, o, 32));
    float sm = 0.f;
    for (int c = lane; c < 512; c += 32) {
      float e = __expf(Lg[r][c] - mx);
      Lg[r][c] = e;
      sm += e;
    }
    #pragma unroll
    for (int o = 16; o > 0; o >>= 1) sm += __shfl_xor(sm, o, 32);
    float inv = 1.f / sm;
    for (int c = lane; c < 512; c += 32) Lg[r][c] *= inv;
  }
  __syncthreads();

  // ---- attn @ V : waves 0/1 each own a 16-col tile of the 32-wide head ----
  if (wave < 2) {
    int nt = wave * 16;
    v8f acc2 = {};
    for (int kst = 0; kst < 16; ++kst) {
      v16bf aP, bV;
      #pragma unroll
      for (int e = 0; e < 16; ++e)
        aP[e] = (bf16)Lg[lane & 15][kst * 32 + kmap(e, lane)];
      #pragma unroll
      for (int e = 0; e < 16; ++e) {
        int k = kst * 32 + kmap(e, lane);
        bV[e] = vh[((size_t)(k * B + b)) * 256 + h * 32 + nt + (lane & 15)];
      }
      acc2 = WMMA_BF16(aP, bV, acc2);
    }
    #pragma unroll
    for (int v = 0; v < 8; ++v) {
      int q = v + ((lane >> 4) << 3);
      if (q < Q)
        out[((size_t)(q * B + b)) * 256 + h * 32 + nt + (lane & 15)] = acc2[v];
    }
  }
}

// ---------------------------------------------------------------------------
// Tiny self-attention (Q=12, hd=32) per (b,h) block — f32 VALU
// ---------------------------------------------------------------------------
__global__ __launch_bounds__(192) void self_attn_kernel(
    const float* __restrict__ qh, const float* __restrict__ kh,
    const float* __restrict__ vh, float* __restrict__ out)
{
  __shared__ float qs[12][32], ks_[12][32], vs[12][32], lg[12][12], pr[12][12];
  const int bh = blockIdx.x;
  const int b = bh >> 3;
  const int h = bh & 7;
  const int tid = threadIdx.x;

  for (int idx = tid; idx < 384; idx += 192) {
    int r = idx >> 5, c = idx & 31;
    size_t o = ((size_t)(r * B + b)) * 256 + h * 32 + c;
    qs[r][c] = qh[o]; ks_[r][c] = kh[o]; vs[r][c] = vh[o];
  }
  __syncthreads();
  if (tid < 144) {
    int qi = tid / 12, ki = tid % 12;
    float s = 0.f;
    #pragma unroll
    for (int d = 0; d < 32; ++d) s += qs[qi][d] * ks_[ki][d];
    lg[qi][ki] = s;
  }
  __syncthreads();
  if (tid < 12) {
    float mx = -1e30f;
    #pragma unroll
    for (int k = 0; k < 12; ++k) mx = fmaxf(mx, lg[tid][k]);
    float sm = 0.f;
    #pragma unroll
    for (int k = 0; k < 12; ++k) { float e = __expf(lg[tid][k] - mx); pr[tid][k] = e; sm += e; }
    float inv = 1.f / sm;
    #pragma unroll
    for (int k = 0; k < 12; ++k) pr[tid][k] *= inv;
  }
  __syncthreads();
  for (int idx = tid; idx < 384; idx += 192) {
    int r = idx >> 5, c = idx & 31;
    float s = 0.f;
    #pragma unroll
    for (int k = 0; k < 12; ++k) s += pr[r][k] * vs[k][c];
    out[((size_t)(r * B + b)) * 256 + h * 32 + c] = s;
  }
}

// ---------------------------------------------------------------------------
// Residual add + LayerNorm (row = 256), optional bf16 mirror
// ---------------------------------------------------------------------------
__global__ __launch_bounds__(256) void ln_residual_kernel(
    const float* __restrict__ x, const float* __restrict__ res,
    const float* __restrict__ g, const float* __restrict__ bt,
    float* __restrict__ out, bf16* __restrict__ out_bf)
{
  __shared__ float red[256];
  __shared__ float bc[2];
  const int r = blockIdx.x, t = threadIdx.x;
  const size_t o = (size_t)r * 256 + t;
  float v = x[o] + res[o];
  red[t] = v;
  __syncthreads();
  for (int s = 128; s > 0; s >>= 1) { if (t < s) red[t] += red[t + s]; __syncthreads(); }
  if (t == 0) bc[0] = red[0] * (1.f / 256.f);
  __syncthreads();
  float d = v - bc[0];
  red[t] = d * d;
  __syncthreads();
  for (int s = 128; s > 0; s >>= 1) { if (t < s) red[t] += red[t + s]; __syncthreads(); }
  if (t == 0) bc[1] = red[0] * (1.f / 256.f);
  __syncthreads();
  float y = d * rsqrtf(bc[1] + 1e-5f) * g[t] + bt[t];
  out[o] = y;
  if (out_bf) out_bf[o] = (bf16)y;
}

// ---------------------------------------------------------------------------
// Elementwise helpers
// ---------------------------------------------------------------------------
__global__ void cast_f32_bf16_kernel(const float* __restrict__ s,
                                     bf16* __restrict__ d, int n)
{
  int i = blockIdx.x * 256 + threadIdx.x;
  if (i < n) d[i] = (bf16)s[i];
}

__global__ void prep_gauss_kernel(const float* __restrict__ gm,
                                  const float* __restrict__ pp,
                                  const float* __restrict__ bb,
                                  float* __restrict__ spp, float* __restrict__ scl)
{
  int idx = blockIdx.x * 256 + threadIdx.x;   // over Q*B*H*3
  if (idx >= QB * H * 3) return;
  int i  = idx % 3;
  int h  = (idx / 3) % H;
  int rb = idx / (3 * H);          // q*B + b
  int q  = rb / B;
  int b  = rb % B;
  float off = tanhf(gm[(size_t)rb * 48 + h * 3 + i]);
  float p   = pp[(size_t)rb * 3 + i] * 8.f;
  float lo  = bb[(size_t)rb * 6 + i] * 8.f;
  float hi  = bb[(size_t)rb * 6 + 3 + i] * 8.f;
  float so0 = -lo * off, so1 = hi * off;
  float vals = fmaxf(so0, so1);
  float sgn  = (so1 > so0) ? 1.f : -1.f;   // argmax (ties -> index 0 -> sign -1)
  size_t o = ((size_t)(b * H + h) * Q + q) * 3 + i;
  spp[o] = p + vals * sgn;
  scl[o] = gm[(size_t)rb * 48 + 24 + h * 3 + i];
}

__global__ void concat_q3_kernel(const float* __restrict__ qc,
                                 const float* __restrict__ qs,
                                 const float* __restrict__ bqs,
                                 bf16* __restrict__ o)
{
  int i = blockIdx.x * 256 + threadIdx.x;
  if (i >= QB * 768) return;
  int r = i / 768, c = i % 768;
  int h = c / 96, j = c % 96;
  size_t base = (size_t)r * 256 + h * 32;
  float v;
  if (j < 32)      v = qc[base + j];
  else if (j < 64) v = qs[base + j - 32];
  else             v = bqs[base + j - 64];
  o[i] = (bf16)v;
}

__global__ void concat_k3_kernel(const float* __restrict__ kc,
                                 const float* __restrict__ kp,
                                 bf16* __restrict__ o)
{
  int i = blockIdx.x * 256 + threadIdx.x;
  if (i >= SB * 768) return;
  int r = i / 768, c = i % 768;
  int h = c / 96, j = c % 96;
  size_t base = (size_t)r * 256 + h * 32;
  float v;
  if (j < 32)      v = kc[base + j] + kp[base + j];
  else if (j < 64) v = kp[base + j - 32];
  else             v = kp[base + j - 64];
  o[i] = (bf16)v;
}

// ---------------------------------------------------------------------------
// Host orchestration
// ---------------------------------------------------------------------------
enum {
  IN_TGT = 0, IN_MEM, IN_POS, IN_QP, IN_BQP, IN_QSE, IN_BQSE, IN_PP, IN_BB,
  W_SA_QC = 9, B_SA_QC, W_SA_PQ, B_SA_PQ, W_SA_BQ, B_SA_BQ,
  W_SA_KC, B_SA_KC, W_SA_PK, B_SA_PK, W_SA_BK, B_SA_BK,
  W_SA_V, B_SA_V,
  W_SA_WQ, B_SA_WQ, W_SA_WK, B_SA_WK, W_SA_WV, B_SA_WV, W_SA_WO, B_SA_WO,
  W_G1, B_G1, W_G2, B_G2, W_G3, B_G3,
  W_CA_QC, B_CA_QC, W_CA_KC, B_CA_KC, W_CA_KPOS, B_CA_KPOS, W_CA_V, B_CA_V,
  W_CA_PQ, B_CA_PQ, W_CA_BQ, B_CA_BQ, W_CA_QS, B_CA_QS, W_CA_BQS, B_CA_BQS,
  W_CA_WQ, B_CA_WQ, W_CA_WK, B_CA_WK, W_CA_WV, B_CA_WV, W_CA_WO, B_CA_WO,
  W_L1, B_L1, W_L2, B_L2,
  G_N1, Bt_N1, G_N2, Bt_N2, G_N3, Bt_N3
};

extern "C" void kernel_launch(void* const* d_in, const int* in_sizes, int n_in,
                              void* d_out, int out_size, void* d_ws, size_t ws_size,
                              hipStream_t stream) {
  (void)in_sizes; (void)n_in; (void)out_size; (void)ws_size;

  char* wp = (char*)d_ws;
  auto allocF = [&](size_t n) { float* r = (float*)wp; wp += ((n * 4 + 255) & ~(size_t)255); return r; };
  auto allocB = [&](size_t n) { bf16*  r = (bf16*)wp;  wp += ((n * 2 + 255) & ~(size_t)255); return r; };
  auto F = [&](int i) { return (const float*)d_in[i]; };

  auto castTo = [&](const float* s, bf16* d, size_t n) {
    cast_f32_bf16_kernel<<<dim3((unsigned)((n + 255) / 256)), 256, 0, stream>>>(s, d, (int)n);
  };
  auto wcast = [&](int idx, size_t n) { bf16* p = allocB(n); castTo(F(idx), p, n); return p; };
  auto gemm = [&](const bf16* A, const bf16* W, const float* bias, float* Cf, bf16* Cb,
                  int M, int N, int K, float alpha, int relu, int accum) {
    dim3 g((unsigned)((N + 63) / 64), (unsigned)((M + 127) / 128));
    gemm_bf16_kernel<<<g, 256, 0, stream>>>(A, W, bias, Cf, Cb, M, N, K, alpha, relu, accum);
  };

  // ---- bf16 weights ----
  bf16* w_sa_qc = wcast(W_SA_QC, 256 * 256);
  bf16* w_sa_pq = wcast(W_SA_PQ, 256 * 256);
  bf16* w_sa_bq = wcast(W_SA_BQ, 256 * 256);
  bf16* w_sa_kc = wcast(W_SA_KC, 256 * 256);
  bf16* w_sa_pk = wcast(W_SA_PK, 256 * 256);
  bf16* w_sa_bk = wcast(W_SA_BK, 256 * 256);
  bf16* w_sa_v  = wcast(W_SA_V,  256 * 256);
  bf16* w_sa_wq = wcast(W_SA_WQ, 256 * 256);
  bf16* w_sa_wk = wcast(W_SA_WK, 256 * 256);
  bf16* w_sa_wv = wcast(W_SA_WV, 256 * 256);
  bf16* w_sa_wo = wcast(W_SA_WO, 256 * 256);
  bf16* w_g1    = wcast(W_G1, 256 * 256);
  bf16* w_g2    = wcast(W_G2, 256 * 256);
  bf16* w_g3    = wcast(W_G3, 256 * 48);
  bf16* w_ca_qc = wcast(W_CA_QC, 256 * 256);
  bf16* w_ca_kc = wcast(W_CA_KC, 256 * 256);
  bf16* w_ca_kp = wcast(W_CA_KPOS, 256 * 256);
  bf16* w_ca_v  = wcast(W_CA_V, 256 * 256);
  bf16* w_ca_pq = wcast(W_CA_PQ, 256 * 256);
  bf16* w_ca_bq = wcast(W_CA_BQ, 256 * 256);
  bf16* w_ca_qs = wcast(W_CA_QS, 256 * 256);
  bf16* w_ca_bqs = wcast(W_CA_BQS, 512 * 256);
  bf16* w_ca_wq = wcast(W_CA_WQ, 768 * 768);
  bf16* w_ca_wk = wcast(W_CA_WK, 768 * 768);
  bf16* w_ca_wv = wcast(W_CA_WV, 256 * 256);
  bf16* w_ca_wo = wcast(W_CA_WO, 256 * 256);
  bf16* w_l1    = wcast(W_L1, 256 * 2048);
  bf16* w_l2    = wcast(W_L2, 2048 * 256);

  // ---- bf16 activations of raw inputs ----
  bf16* tgt_bf  = allocB((size_t)QB * 256); castTo(F(IN_TGT),  tgt_bf,  (size_t)QB * 256);
  bf16* qp_bf   = allocB((size_t)QB * 256); castTo(F(IN_QP),   qp_bf,   (size_t)QB * 256);
  bf16* bqp_bf  = allocB((size_t)QB * 256); castTo(F(IN_BQP),  bqp_bf,  (size_t)QB * 256);
  bf16* mem_bf  = allocB((size_t)SB * 256); castTo(F(IN_MEM),  mem_bf,  (size_t)SB * 256);
  bf16* pos_bf  = allocB((size_t)SB * 256); castTo(F(IN_POS),  pos_bf,  (size_t)SB * 256);
  bf16* qse_bf  = allocB((size_t)QB * 256); castTo(F(IN_QSE),  qse_bf,  (size_t)QB * 256);
  bf16* bqse_bf = allocB((size_t)QB * 512); castTo(F(IN_BQSE), bqse_bf, (size_t)QB * 512);

  // ================= self attention =================
  float* q_sa = allocF((size_t)QB * 256);
  float* k_sa = allocF((size_t)QB * 256);
  float* v_sa = allocF((size_t)QB * 256);
  gemm(tgt_bf, w_sa_qc, F(B_SA_QC), q_sa, nullptr, QB, 256, 256, 1.f, 0, 0);
  gemm(qp_bf,  w_sa_pq, F(B_SA_PQ), q_sa, nullptr, QB, 256, 256, 1.f, 0, 1);
  gemm(bqp_bf, w_sa_bq, F(B_SA_BQ), q_sa, nullptr, QB, 256, 256, 1.f, 0, 1);
  gemm(tgt_bf, w_sa_kc, F(B_SA_KC), k_sa, nullptr, QB, 256, 256, 1.f, 0, 0);
  gemm(qp_bf,  w_sa_pk, F(B_SA_PK), k_sa, nullptr, QB, 256, 256, 1.f, 0, 1);
  gemm(bqp_bf, w_sa_bk, F(B_SA_BK), k_sa, nullptr, QB, 256, 256, 1.f, 0, 1);
  gemm(tgt_bf, w_sa_v,  F(B_SA_V),  v_sa, nullptr, QB, 256, 256, 1.f, 0, 0);

  bf16* q_sa_bf = allocB((size_t)QB * 256); castTo(q_sa, q_sa_bf, (size_t)QB * 256);
  bf16* k_sa_bf = allocB((size_t)QB * 256); castTo(k_sa, k_sa_bf, (size_t)QB * 256);
  bf16* v_sa_bf = allocB((size_t)QB * 256); castTo(v_sa, v_sa_bf, (size_t)QB * 256);

  float* qh_sa = allocF((size_t)QB * 256);
  float* kh_sa = allocF((size_t)QB * 256);
  float* vh_sa = allocF((size_t)QB * 256);
  const float inv_sqrt_hd = 0.17677669529663687f;   // 32^-0.5
  gemm(q_sa_bf, w_sa_wq, F(B_SA_WQ), qh_sa, nullptr, QB, 256, 256, inv_sqrt_hd, 0, 0);
  gemm(k_sa_bf, w_sa_wk, F(B_SA_WK), kh_sa, nullptr, QB, 256, 256, 1.f, 0, 0);
  gemm(v_sa_bf, w_sa_wv, F(B_SA_WV), vh_sa, nullptr, QB, 256, 256, 1.f, 0, 0);

  float* sa_out = allocF((size_t)QB * 256);
  self_attn_kernel<<<B * H, 192, 0, stream>>>(qh_sa, kh_sa, vh_sa, sa_out);

  bf16* sa_out_bf = allocB((size_t)QB * 256); castTo(sa_out, sa_out_bf, (size_t)QB * 256);
  float* sa_proj = allocF((size_t)QB * 256);
  gemm(sa_out_bf, w_sa_wo, F(B_SA_WO), sa_proj, nullptr, QB, 256, 256, 1.f, 0, 0);

  float* tgt2    = allocF((size_t)QB * 256);
  bf16*  tgt2_bf = allocB((size_t)QB * 256);
  ln_residual_kernel<<<QB, 256, 0, stream>>>(sa_proj, F(IN_TGT), F(G_N1), F(Bt_N1), tgt2, tgt2_bf);

  // ================= gaussian MLP =================
  bf16* h1_bf = allocB((size_t)QB * 256);
  bf16* h2_bf = allocB((size_t)QB * 256);
  float* gm   = allocF((size_t)QB * 48);
  gemm(tgt2_bf, w_g1, F(B_G1), nullptr, h1_bf, QB, 256, 256, 1.f, 1, 0);
  gemm(h1_bf,   w_g2, F(B_G2), nullptr, h2_bf, QB, 256, 256, 1.f, 1, 0);
  gemm(h2_bf,   w_g3, F(B_G3), gm, nullptr,    QB, 48, 256, 1.f, 0, 0);

  float* spp = allocF((size_t)B * H * Q * 3);
  float* scl = allocF((size_t)B * H * Q * 3);
  prep_gauss_kernel<<<(QB * H * 3 + 255) / 256, 256, 0, stream>>>(gm, F(IN_PP), F(IN_BB), spp, scl);

  // ================= cross attention projections =================
  float* qc_sum = allocF((size_t)QB * 256);
  gemm(tgt2_bf, w_ca_qc, F(B_CA_QC), qc_sum, nullptr, QB, 256, 256, 1.f, 0, 0);
  gemm(qp_bf,   w_ca_pq, F(B_CA_PQ), qc_sum, nullptr, QB, 256, 256, 1.f, 0, 1);
  gemm(bqp_bf,  w_ca_bq, F(B_CA_BQ), qc_sum, nullptr, QB, 256, 256, 1.f, 0, 1);

  float* qse_p  = allocF((size_t)QB * 256);
  float* bqse_p = allocF((size_t)QB * 256);
  gemm(qse_bf,  w_ca_qs,  F(B_CA_QS),  qse_p,  nullptr, QB, 256, 256, 1.f, 0, 0);
  gemm(bqse_bf, w_ca_bqs, F(B_CA_BQS), bqse_p, nullptr, QB, 256, 512, 1.f, 0, 0);

  bf16* q3_bf = allocB((size_t)QB * 768);
  concat_q3_kernel<<<(QB * 768 + 255) / 256, 256, 0, stream>>>(qc_sum, qse_p, bqse_p, q3_bf);

  bf16* qh3_bf = allocB((size_t)QB * 768);
  const float inv_sqrt_hd3 = 0.10206207261596575f;  // 96^-0.5
  gemm(q3_bf, w_ca_wq, F(B_CA_WQ), nullptr, qh3_bf, QB, 768, 768, inv_sqrt_hd3, 0, 0);

  float* kc = allocF((size_t)SB * 256);
  float* kp = allocF((size_t)SB * 256);
  bf16*  v_ca_bf = allocB((size_t)SB * 256);
  gemm(mem_bf, w_ca_kc, F(B_CA_KC),   kc, nullptr, SB, 256, 256, 1.f, 0, 0);
  gemm(pos_bf, w_ca_kp, F(B_CA_KPOS), kp, nullptr, SB, 256, 256, 1.f, 0, 0);
  gemm(mem_bf, w_ca_v,  F(B_CA_V), nullptr, v_ca_bf, SB, 256, 256, 1.f, 0, 0);

  bf16* k3_bf = allocB((size_t)SB * 768);
  concat_k3_kernel<<<(unsigned)(((size_t)SB * 768 + 255) / 256), 256, 0, stream>>>(kc, kp, k3_bf);

  bf16* kh3_bf = allocB((size_t)SB * 768);
  gemm(k3_bf, w_ca_wk, F(B_CA_WK), nullptr, kh3_bf, SB, 768, 768, 1.f, 0, 0);

  bf16* vh_bf = allocB((size_t)SB * 256);
  gemm(v_ca_bf, w_ca_wv, F(B_CA_WV), nullptr, vh_bf, SB, 256, 256, 1.f, 0, 0);

  float* ca_out = allocF((size_t)QB * 256);
  cross_attn_kernel<<<B * H, 128, 0, stream>>>(qh3_bf, kh3_bf, vh_bf, spp, scl, ca_out);

  bf16* ca_out_bf = allocB((size_t)QB * 256); castTo(ca_out, ca_out_bf, (size_t)QB * 256);
  float* ca_proj = allocF((size_t)QB * 256);
  gemm(ca_out_bf, w_ca_wo, F(B_CA_WO), ca_proj, nullptr, QB, 256, 256, 1.f, 0, 0);

  float* tgt3    = allocF((size_t)QB * 256);
  bf16*  tgt3_bf = allocB((size_t)QB * 256);
  ln_residual_kernel<<<QB, 256, 0, stream>>>(ca_proj, tgt2, F(G_N2), F(Bt_N2), tgt3, tgt3_bf);

  // ================= FFN =================
  bf16* ff1_bf = allocB((size_t)QB * DFF);
  gemm(tgt3_bf, w_l1, F(B_L1), nullptr, ff1_bf, QB, DFF, 256, 1.f, 1, 0);
  float* ff2 = allocF((size_t)QB * 256);
  gemm(ff1_bf, w_l2, F(B_L2), ff2, nullptr, QB, 256, DFF, 1.f, 0, 0);

  ln_residual_kernel<<<QB, 256, 0, stream>>>(ff2, tgt3, F(G_N3), F(Bt_N3), (float*)d_out, nullptr);
}
